// NFLPredictor_56040733278779
// MI455X (gfx1250) — matrix operand order repaired and checked by reading, not compiled, not run on hardware
//
#include <hip/hip_runtime.h>

typedef __attribute__((ext_vector_type(16))) _Float16 v16h;
typedef __attribute__((ext_vector_type(8)))  _Float16 v8h;
typedef __attribute__((ext_vector_type(8)))  float    v8f;

#define N_SEQ   22528     // B*P = 1024*22
#define T_ENC   20
#define T_DEC   25
#define IN_DIM  33
#define HID     64
#define WPB     8         // waves per block (256 threads)

// ---- encoder shared-memory layout (bytes) ----
#define ENC_OFF_WIH   0                         // 256*64 halfs  (K padded 33->64)
#define ENC_OFF_WHH   (256*64*2)                // 256*64 halfs
#define ENC_OFF_B     (ENC_OFF_WHH + 256*64*2)  // 256 floats
#define ENC_OFF_H     (ENC_OFF_B + 256*4)       // 8 * 16*64 halfs
#define ENC_SHM       (ENC_OFF_H + WPB*16*64*2) // 82944 B

// ---- decoder shared-memory layout (bytes) ----
#define DEC_OFF_WHH   0                          // 256*64 halfs
#define DEC_OFF_WIH   (256*64*2)                 // 256*2  floats
#define DEC_OFF_B     (DEC_OFF_WIH + 512*4)      // 256 floats
#define DEC_OFF_FCW   (DEC_OFF_B + 256*4)        // 2*64 floats
#define DEC_OFF_FCB   (DEC_OFF_FCW + 128*4)      // 2 floats (padded to 32B)
#define DEC_OFF_H     (DEC_OFF_FCB + 32)         // 8 * 16*64 halfs
#define DEC_OFF_X     (DEC_OFF_H + WPB*16*64*2)  // 8 * 16*2 floats
#define DEC_SHM       (DEC_OFF_X + WPB*32*4)

__device__ __forceinline__ float sigmoidf_(float x) {
  return 1.0f / (1.0f + __expf(-x));
}
// tanh via exp: correct +/-1 saturation at extremes, single v_exp_f32
__device__ __forceinline__ float tanhf_(float x) {
  return 1.0f - 2.0f / (1.0f + __expf(2.0f * x));
}

__device__ __forceinline__ v16h hcat(v8h lo, v8h hi) {
  v16h r;
#pragma unroll
  for (int i = 0; i < 8; ++i) { r[i] = lo[i]; r[8 + i] = hi[i]; }
  return r;
}

// =====================================================================
// Encoder LSTM: one wave per 16-sequence tile, 20 recurrent steps.
// gates[16x256] = x_pad[16x64] @ WihT + h[16x64] @ WhhT + b   (WMMA f16)
// WMMA chains start from a zero accumulator (rematerializable); bias is
// added post-chain so nothing loop-invariant-but-fat stays live.
// =====================================================================
__global__ void enc_lstm_kernel(const float* __restrict__ x,
                                const float* __restrict__ Wih,   // 256 x 33
                                const float* __restrict__ Whh,   // 256 x 64
                                const float* __restrict__ bih,
                                const float* __restrict__ bhh,
                                float* __restrict__ hn,          // n x 64
                                float* __restrict__ cn)          // n x 64
{
  extern __shared__ char smem[];
  _Float16* sWih = (_Float16*)(smem + ENC_OFF_WIH);
  _Float16* sWhh = (_Float16*)(smem + ENC_OFF_WHH);
  float*    sB   = (float*)   (smem + ENC_OFF_B);

  const int tid  = threadIdx.x;
  const int lane = tid & 31;
  const int wave = tid >> 5;

  // cooperative weight load + f32 -> f16 conversion (Wih K-padded to 64)
  for (int i = tid; i < 256 * 64; i += 256) {
    int r = i >> 6, k = i & 63;
    sWih[i] = (k < IN_DIM) ? (_Float16)Wih[r * IN_DIM + k] : (_Float16)0.f;
    sWhh[i] = (_Float16)Whh[i];
  }
  if (tid < 256) sB[tid] = bih[tid] + bhh[tid];
  __syncthreads();

  const int seq0  = (blockIdx.x * WPB + wave) * 16;
  const int rowA  = lane & 15;      // A-frag M row this lane feeds
  const int hiA   = lane >> 4;      // lane half selects K interleave
  const int kloA  = hiA * 8;
  const int colN  = lane & 15;      // C/D N column within 16-tile
  const int khB   = hiA * 8;        // B-frag K half offset (halfs)
  const int rowM0 = hiA * 8;        // C/D M base

  // single per-lane LDS bases; everything else is an immediate offset
  const char* wBase = smem + (colN * 128 + khB * 2);           // weights
  const char* bBase = smem + (ENC_OFF_B + colN * 4);           // bias
  char*       hA    = smem + (ENC_OFF_H + wave * 2048 + rowA * 128 + kloA * 2);
  char*       hS    = smem + (ENC_OFF_H + wave * 2048 + rowM0 * 128 + colN * 2);
  _Float16*   sH    = (_Float16*)(smem + ENC_OFF_H + wave * 2048);

  for (int i = lane; i < 16 * 64; i += 32) sH[i] = (_Float16)0.f;

  v8f c[4];
#pragma unroll
  for (int jt = 0; jt < 4; ++jt) c[jt] = (v8f)(0.f);

#pragma unroll 1
  for (int t = 0; t < T_ENC; ++t) {
    // ---- A fragments for x_t (K padded 33 -> 64, zeros beyond) ----
    v16h ax[2];
    {
      const long base = ((long)(seq0 + rowA) * T_ENC + t) * IN_DIM;
      if (t + 1 < T_ENC) __builtin_prefetch(&x[base + IN_DIM], 0, 1);
#pragma unroll
      for (int kc = 0; kc < 2; ++kc) {
        v16h a;
#pragma unroll
        for (int e = 0; e < 8; ++e) {
          int k0 = kc * 32 + kloA + e;
          int k1 = kc * 32 + 16 + kloA + e;
          a[e]     = (k0 < IN_DIM) ? (_Float16)x[base + k0] : (_Float16)0.f;
          a[8 + e] = (k1 < IN_DIM) ? (_Float16)x[base + k1] : (_Float16)0.f;
        }
        ax[kc] = a;
      }
    }
    // ---- A fragments for h (per-wave LDS tile, imm offsets) ----
    v16h ah[2];
#pragma unroll
    for (int kc = 0; kc < 2; ++kc)
      ah[kc] = hcat(*(const v8h*)(hA + kc * 64), *(const v8h*)(hA + kc * 64 + 32));

#pragma unroll
    for (int jt = 0; jt < 4; ++jt) {
      v8f acc[4];
      float bb[4];
#pragma unroll
      for (int gg = 0; gg < 4; ++gg) {               // i, f, g, o
        const int ct = gg * 4 + jt;                  // 16-wide gate column tile
        bb[gg] = *(const float*)(bBase + ct * 64);
        v8f a = (v8f)(0.f);                          // rematerializable C
#pragma unroll
        for (int kc = 0; kc < 2; ++kc) {             // x part
          v16h bf = hcat(*(const v8h*)(wBase + ENC_OFF_WIH + ct * 2048 + kc * 64), *(const v8h*)(wBase + ENC_OFF_WIH + ct * 2048 + kc * 64 + 32));
          a = __builtin_amdgcn_wmma_f32_16x16x32_f16(false, ax[kc], false, bf,
                                                     (short)0, a, false, false);
        }
#pragma unroll
        for (int kc = 0; kc < 2; ++kc) {             // h part
          v16h bf = hcat(*(const v8h*)(wBase + ENC_OFF_WHH + ct * 2048 + kc * 64), *(const v8h*)(wBase + ENC_OFF_WHH + ct * 2048 + kc * 64 + 32));
          a = __builtin_amdgcn_wmma_f32_16x16x32_f16(false, ah[kc], false, bf,
                                                     (short)0, a, false, false);
        }
        acc[gg] = a;
      }
      // ---- cell update for hidden columns jt*16 + colN ----
#pragma unroll
      for (int r = 0; r < 8; ++r) {
        float iv = sigmoidf_(acc[0][r] + bb[0]);
        float fv = sigmoidf_(acc[1][r] + bb[1]);
        float gv = tanhf_(acc[2][r] + bb[2]);
        float ov = sigmoidf_(acc[3][r] + bb[3]);
        float cv = fv * c[jt][r] + iv * gv;
        c[jt][r] = cv;
        float hv = ov * tanhf_(cv);
        *(_Float16*)(hS + r * 128 + jt * 32) = (_Float16)hv;
      }
    }
  }

  // ---- final-state writeback (out of hot loop) ----
  for (int i = lane; i < 16 * 64; i += 32)
    hn[(long)(seq0 + (i >> 6)) * HID + (i & 63)] = (float)sH[i];
#pragma unroll
  for (int jt = 0; jt < 4; ++jt)
#pragma unroll
    for (int r = 0; r < 8; ++r)
      cn[(long)(seq0 + rowM0 + r) * HID + jt * 16 + colN] = c[jt][r];
}

// =====================================================================
// Attention over P=22 players per batch + residual + LayerNorm.
// Tiny (22x22x64 per block) -> VALU with LDS staging.
// =====================================================================
__global__ void attn_ln_kernel(const float* __restrict__ hn,
                               const float* __restrict__ qW, const float* __restrict__ qb,
                               const float* __restrict__ kW, const float* __restrict__ kb,
                               const float* __restrict__ vW, const float* __restrict__ vb,
                               const float* __restrict__ lng, const float* __restrict__ lnb,
                               float* __restrict__ y)
{
  __shared__ float sh[22 * 64];
  __shared__ float sQ[22 * 64];
  __shared__ float sK[22 * 64];
  __shared__ float sV[22 * 64];
  __shared__ float sS[22 * 22];

  const int b   = blockIdx.x;
  const int tid = threadIdx.x;          // 64 threads

  for (int i = tid; i < 22 * 64; i += 64) sh[i] = hn[(long)b * 22 * 64 + i];
  __syncthreads();

  {                                     // Q,K,V projections; thread = column j
    const int j = tid;
    for (int p = 0; p < 22; ++p) {
      float aq = 0.f, ak = 0.f, av = 0.f;
      for (int k = 0; k < 64; ++k) {
        float h = sh[p * 64 + k];
        aq += h * qW[j * 64 + k];
        ak += h * kW[j * 64 + k];
        av += h * vW[j * 64 + k];
      }
      sQ[p * 64 + j] = aq + qb[j];
      sK[p * 64 + j] = ak + kb[j];
      sV[p * 64 + j] = av + vb[j];
    }
  }
  __syncthreads();

  for (int idx = tid; idx < 22 * 22; idx += 64) {
    int p = idx / 22, q = idx % 22;
    float s = 0.f;
    for (int k = 0; k < 64; ++k) s += sQ[p * 64 + k] * sK[q * 64 + k];
    sS[idx] = s * 0.125f;               // 1/sqrt(64)
  }
  __syncthreads();

  if (tid < 22) {                       // softmax over row tid
    float m = -1e30f;
    for (int q = 0; q < 22; ++q) m = fmaxf(m, sS[tid * 22 + q]);
    float sum = 0.f;
    for (int q = 0; q < 22; ++q) sum += __expf(sS[tid * 22 + q] - m);
    float inv = 1.f / sum;
    for (int q = 0; q < 22; ++q)
      sS[tid * 22 + q] = __expf(sS[tid * 22 + q] - m) * inv;
  }
  __syncthreads();

  {                                     // ctx + residual into sQ (reuse)
    const int j = tid;
    for (int p = 0; p < 22; ++p) {
      float cx = 0.f;
      for (int q = 0; q < 22; ++q) cx += sS[p * 22 + q] * sV[q * 64 + j];
      sQ[p * 64 + j] = sh[p * 64 + j] + cx;
    }
  }
  __syncthreads();

  if (tid < 22) {                       // layernorm over row tid
    const int p = tid;
    float mu = 0.f;
    for (int j = 0; j < 64; ++j) mu += sQ[p * 64 + j];
    mu *= (1.f / 64.f);
    float var = 0.f;
    for (int j = 0; j < 64; ++j) { float d = sQ[p * 64 + j] - mu; var += d * d; }
    var *= (1.f / 64.f);
    float rstd = rsqrtf(var + 1e-5f);
    for (int j = 0; j < 64; ++j)
      y[((long)b * 22 + p) * 64 + j] = (sQ[p * 64 + j] - mu) * rstd * lng[j] + lnb[j];
  }
}

// =====================================================================
// Decoder LSTM: h-part via WMMA from zero accumulator; bias + K=2 input
// term fused into the nonlinearity input. FC(64->2) + feedback per-lane.
// =====================================================================
__global__ void dec_lstm_kernel(const float* __restrict__ x,     // for inp0
                                const float* __restrict__ y0,    // n x 64
                                const float* __restrict__ c0,    // n x 64
                                const float* __restrict__ Wih,   // 256 x 2
                                const float* __restrict__ Whh,   // 256 x 64
                                const float* __restrict__ bih,
                                const float* __restrict__ bhh,
                                const float* __restrict__ fcW,   // 2 x 64
                                const float* __restrict__ fcb,   // 2
                                float* __restrict__ out)         // n x 25 x 2
{
  extern __shared__ char smem[];
  _Float16* sWhh = (_Float16*)(smem + DEC_OFF_WHH);
  float*    sWih = (float*)   (smem + DEC_OFF_WIH);
  float*    sB   = (float*)   (smem + DEC_OFF_B);
  float*    sFcW = (float*)   (smem + DEC_OFF_FCW);
  float*    sFcb = (float*)   (smem + DEC_OFF_FCB);

  const int tid  = threadIdx.x;
  const int lane = tid & 31;
  const int wave = tid >> 5;

  for (int i = tid; i < 256 * 64; i += 256) sWhh[i] = (_Float16)Whh[i];
  if (tid < 256) sB[tid] = bih[tid] + bhh[tid];
  for (int i = tid; i < 512; i += 256) sWih[i] = Wih[i];
  if (tid < 128) sFcW[tid] = fcW[tid];
  if (tid < 2)   sFcb[tid] = fcb[tid];
  __syncthreads();

  const int seq0  = (blockIdx.x * WPB + wave) * 16;
  const int rowA  = lane & 15;
  const int hiA   = lane >> 4;
  const int kloA  = hiA * 8;
  const int colN  = lane & 15;
  const int khB   = hiA * 8;
  const int rowM0 = hiA * 8;

  const char* wBase = smem + (DEC_OFF_WHH + colN * 128 + khB * 2);
  const char* bBase = smem + (DEC_OFF_B + colN * 4);
  const char* iBase = smem + (DEC_OFF_WIH + colN * 8);
  char*       hA    = smem + (DEC_OFF_H + wave * 2048 + rowA * 128 + kloA * 2);
  char*       hS    = smem + (DEC_OFF_H + wave * 2048 + rowM0 * 128 + colN * 2);
  _Float16*   sH    = (_Float16*)(smem + DEC_OFF_H + wave * 2048);
  float*      sX    = (float*)   (smem + DEC_OFF_X + wave * 128);
  const char* xBase = smem + (DEC_OFF_X + wave * 128 + rowM0 * 8);

  // h0 = y (layernormed), c0 = encoder final cell, inp0 = x[:, -1, :2]
  for (int i = lane; i < 16 * 64; i += 32)
    sH[i] = (_Float16)y0[(long)(seq0 + (i >> 6)) * HID + (i & 63)];

  v8f c[4];
#pragma unroll
  for (int jt = 0; jt < 4; ++jt)
#pragma unroll
    for (int r = 0; r < 8; ++r)
      c[jt][r] = c0[(long)(seq0 + rowM0 + r) * HID + jt * 16 + colN];

  if (lane < 16) {
    const long base = ((long)(seq0 + lane) * T_ENC + (T_ENC - 1)) * IN_DIM;
    sX[lane * 2 + 0] = x[base + 0];
    sX[lane * 2 + 1] = x[base + 1];
  }

  const int prow = lane & 15;           // FC mapping: lane = (row, out-dim)
  const int pd   = lane >> 4;
  float* outp = out + ((long)(seq0 + prow) * T_DEC) * 2 + pd;

#pragma unroll 1
  for (int s = 0; s < T_DEC; ++s) {
    v16h ah[2];
#pragma unroll
    for (int kc = 0; kc < 2; ++kc)
      ah[kc] = hcat(*(const v8h*)(hA + kc * 64), *(const v8h*)(hA + kc * 64 + 32));

#pragma unroll
    for (int jt = 0; jt < 4; ++jt) {
      v8f acc[4];
      float bb[4], w0[4], w1[4];
#pragma unroll
      for (int gg = 0; gg < 4; ++gg) {
        const int ct = gg * 4 + jt;
        bb[gg] = *(const float*)(bBase + ct * 64);
        w0[gg] = *(const float*)(iBase + ct * 128);
        w1[gg] = *(const float*)(iBase + ct * 128 + 4);
        v8f a = (v8f)(0.f);                          // rematerializable C
#pragma unroll
        for (int kc = 0; kc < 2; ++kc) {
          v16h bf = hcat(*(const v8h*)(wBase + ct * 2048 + kc * 64), *(const v8h*)(wBase + ct * 2048 + kc * 64 + 32));
          a = __builtin_amdgcn_wmma_f32_16x16x32_f16(false, ah[kc], false, bf,
                                                     (short)0, a, false, false);
        }
        acc[gg] = a;
      }
#pragma unroll
      for (int r = 0; r < 8; ++r) {
        float x0 = *(const float*)(xBase + r * 8);
        float x1 = *(const float*)(xBase + r * 8 + 4);
        float iv = sigmoidf_(acc[0][r] + bb[0] + x0 * w0[0] + x1 * w1[0]);
        float fv = sigmoidf_(acc[1][r] + bb[1] + x0 * w0[1] + x1 * w1[1]);
        float gv = tanhf_(acc[2][r] + bb[2] + x0 * w0[2] + x1 * w1[2]);
        float ov = sigmoidf_(acc[3][r] + bb[3] + x0 * w0[3] + x1 * w1[3]);
        float cv = fv * c[jt][r] + iv * gv;
        c[jt][r] = cv;
        float hv = ov * tanhf_(cv);
        *(_Float16*)(hS + r * 128 + jt * 32) = (_Float16)hv;
      }
    }

    // FC 64->2 + feedback; vectorized LDS reads of the fresh h row
    {
      float acc = sFcb[pd];
#pragma unroll
      for (int kb = 0; kb < 8; ++kb) {
        v8h hh = *(const v8h*)(sH + prow * 64 + kb * 8);
#pragma unroll
        for (int e = 0; e < 8; ++e)
          acc += (float)hh[e] * sFcW[pd * 64 + kb * 8 + e];
      }
      outp[s * 2] = acc;
      sX[prow * 2 + pd] = acc;
    }
  }
}

extern "C" void kernel_launch(void* const* d_in, const int* in_sizes, int n_in,
                              void* d_out, int out_size, void* d_ws, size_t ws_size,
                              hipStream_t stream) {
  (void)in_sizes; (void)n_in; (void)out_size; (void)ws_size;
  const float* x        = (const float*)d_in[0];
  const float* enc_Wih  = (const float*)d_in[1];
  const float* enc_Whh  = (const float*)d_in[2];
  const float* enc_bih  = (const float*)d_in[3];
  const float* enc_bhh  = (const float*)d_in[4];
  const float* q_W      = (const float*)d_in[5];
  const float* q_b      = (const float*)d_in[6];
  const float* k_W      = (const float*)d_in[7];
  const float* k_b      = (const float*)d_in[8];
  const float* v_W      = (const float*)d_in[9];
  const float* v_b      = (const float*)d_in[10];
  const float* ln_g     = (const float*)d_in[11];
  const float* ln_b     = (const float*)d_in[12];
  const float* dec_Wih  = (const float*)d_in[13];
  const float* dec_Whh  = (const float*)d_in[14];
  const float* dec_bih  = (const float*)d_in[15];
  const float* dec_bhh  = (const float*)d_in[16];
  const float* fc_W     = (const float*)d_in[17];
  const float* fc_b     = (const float*)d_in[18];

  const long N64 = (long)N_SEQ * HID;
  float* ws  = (float*)d_ws;
  float* hn  = ws;               // n x 64
  float* cn  = ws + N64;         // n x 64
  float* yb  = ws + 2 * N64;     // n x 64

  hipFuncSetAttribute(reinterpret_cast<const void*>(&enc_lstm_kernel),
                      hipFuncAttributeMaxDynamicSharedMemorySize, ENC_SHM);
  hipFuncSetAttribute(reinterpret_cast<const void*>(&dec_lstm_kernel),
                      hipFuncAttributeMaxDynamicSharedMemorySize, DEC_SHM);

  const int nblocks = N_SEQ / (16 * WPB);   // 176
  enc_lstm_kernel<<<nblocks, 32 * WPB, ENC_SHM, stream>>>(
      x, enc_Wih, enc_Whh, enc_bih, enc_bhh, hn, cn);
  attn_ln_kernel<<<1024, 64, 0, stream>>>(
      hn, q_W, q_b, k_W, k_b, v_W, v_b, ln_g, ln_b, yb);
  dec_lstm_kernel<<<nblocks, 32 * WPB, DEC_SHM, stream>>>(
      x, yb, cn, dec_Wih, dec_Whh, dec_bih, dec_bhh, fc_W, fc_b, (float*)d_out);
}
